// DenseGATLayer_4492535792512
// MI455X (gfx1250) — matrix operand order repaired and checked by reading, not compiled, not run on hardware
//
#include <hip/hip_runtime.h>
#include <hip/hip_fp16.h>

#define B_  16
#define N_  1024
#define IN_ 256
#define HD_ 256
#define H_  4
#define LOG2E 1.44269504088896340736f

typedef __attribute__((ext_vector_type(16))) _Float16 v16h;
typedef __attribute__((ext_vector_type(8)))  float    v8f;

__device__ __forceinline__ v8f wmma16(v16h a, v16h b, v8f c) {
  return __builtin_amdgcn_wmma_f32_16x16x32_f16(false, a, false, b, (short)0, c, false, false);
}

struct V16U { union { v16h v; uint4 u[2]; }; };

// A-fragment (16x32 f16): lane<16 -> row row16+lane, K {0..7,16..23}; lane>=16 -> K {8..15,24..31}
__device__ __forceinline__ v16h lds_a_frag(const __half* p, int pitchH, int row16, int kb) {
  int lane = threadIdx.x & 31;
  int l16 = lane & 15, lh = lane >> 4;
  const char* a = (const char*)(p + (size_t)(row16 + l16) * pitchH + kb + lh * 8);
  V16U r;
  r.u[0] = *(const uint4*)a;
  r.u[1] = *(const uint4*)(a + 32);
  return r.v;
}

// B-fragment (32x16 f16): lane gives column, lane<16 -> K {0..7,16..23}; lane>=16 -> K {8..15,24..31} (source transposed in LDS)
__device__ __forceinline__ v16h lds_b_frag(const __half* p, int pitchH, int col16, int kb) {
  int lane = threadIdx.x & 31;
  int l16 = lane & 15, lh = lane >> 4;
  const char* a = (const char*)(p + (size_t)(col16 + l16) * pitchH + kb + lh * 8);
  V16U r;
  r.u[0] = *(const uint4*)a;
  r.u[1] = *(const uint4*)(a + 32);
  return r.v;
}

__device__ __forceinline__ void atomicMaxF(float* addr, float v) {
  if (v >= 0.f) atomicMax((int*)addr, __float_as_int(v));
  else          atomicMin((unsigned int*)addr, __float_as_uint(v));
}

__global__ void kInit(float* Mbuf) {
  if (threadIdx.x < B_ * H_) Mbuf[threadIdx.x] = 0.f;
}

// -------- Kernel A: h = x @ W, f32 -> f16 WMMA GEMM, writes h as f16 --------
__global__ __launch_bounds__(256) void kA_proj(const float* __restrict__ x,
                                               const float* __restrict__ Wg,
                                               __half* __restrict__ hF) {
  __shared__ __half xS[128 * 40];   // [m][k], pitch 40 halves
  __shared__ __half wT[128 * 40];   // [n][k] transposed, pitch 40
  int t = threadIdx.x;
  int n0 = blockIdx.x * 128;
  int m0 = blockIdx.y * 128;
  int w = t >> 5, lane = t & 31, l16 = lane & 15, lh = lane >> 4;
  int mbase = (w & 3) * 32, nbase = (w >> 2) * 64;
  v8f acc[2][4];
#pragma unroll
  for (int a = 0; a < 2; ++a)
#pragma unroll
    for (int bq = 0; bq < 4; ++bq)
#pragma unroll
      for (int e = 0; e < 8; ++e) acc[a][bq][e] = 0.f;

  for (int kk = 0; kk < 8; ++kk) {
    int k0 = kk * 32;
    __syncthreads();
    { // stage x tile -> f16 (b128 stores)
      int m = t >> 1, kb = (t & 1) * 16;
      const float4* xp = (const float4*)(x + (size_t)(m0 + m) * IN_ + k0 + kb);
      union { __half h[16]; uint4 u[2]; } tv;
#pragma unroll
      for (int c = 0; c < 4; ++c) {
        float4 f = xp[c];
        tv.h[c*4+0] = __float2half(f.x); tv.h[c*4+1] = __float2half(f.y);
        tv.h[c*4+2] = __float2half(f.z); tv.h[c*4+3] = __float2half(f.w);
      }
      uint4* dp = (uint4*)(xS + m * 40 + kb);
      dp[0] = tv.u[0]; dp[1] = tv.u[1];
    }
    { // stage W tile transposed, k-pairs packed -> b32 stores
      int k2 = (t & 15) * 2, nb = (t >> 4) * 8;
      const float4* wp0 = (const float4*)(Wg + (size_t)(k0 + k2)     * HD_ + n0 + nb);
      const float4* wp1 = (const float4*)(Wg + (size_t)(k0 + k2 + 1) * HD_ + n0 + nb);
#pragma unroll
      for (int c = 0; c < 2; ++c) {
        float4 f0 = wp0[c];
        float4 f1 = wp1[c];
        int nbc = nb + c * 4;
        *(__half2*)(wT + (nbc + 0) * 40 + k2) = __halves2half2(__float2half(f0.x), __float2half(f1.x));
        *(__half2*)(wT + (nbc + 1) * 40 + k2) = __halves2half2(__float2half(f0.y), __float2half(f1.y));
        *(__half2*)(wT + (nbc + 2) * 40 + k2) = __halves2half2(__float2half(f0.z), __float2half(f1.z));
        *(__half2*)(wT + (nbc + 3) * 40 + k2) = __halves2half2(__float2half(f0.w), __float2half(f1.w));
      }
    }
    __syncthreads();
    v16h a0 = lds_a_frag(xS, 40, mbase, 0);
    v16h a1 = lds_a_frag(xS, 40, mbase + 16, 0);
#pragma unroll
    for (int nt = 0; nt < 4; ++nt) {
      v16h bf = lds_b_frag(wT, 40, nbase + nt * 16, 0);
      acc[0][nt] = wmma16(a0, bf, acc[0][nt]);
      acc[1][nt] = wmma16(a1, bf, acc[1][nt]);
    }
  }
#pragma unroll
  for (int mt = 0; mt < 2; ++mt)
#pragma unroll
    for (int nt = 0; nt < 4; ++nt)
#pragma unroll
      for (int v = 0; v < 8; ++v) {
        int row = m0 + mbase + mt * 16 + v + lh * 8;
        int col = n0 + nbase + nt * 16 + l16;
        hF[(size_t)row * HD_ + col] = __float2half(acc[mt][nt][v]);
      }
}

// -------- Kernel B: src/dst scores + per-(b,h) masked max of dst --------
__global__ __launch_bounds__(256) void kB_scores(const __half* __restrict__ hF,
                                                 const float* __restrict__ a_src,
                                                 const float* __restrict__ a_dst,
                                                 const int* __restrict__ mask,
                                                 float* __restrict__ srcB,
                                                 float* __restrict__ dstB,
                                                 float* __restrict__ Mbuf) {
  int idx = blockIdx.x * 256 + threadIdx.x;   // (b,n,h) flat, 65536 total
  int hh = idx & 3, nn = (idx >> 2) & (N_ - 1), bb = idx >> 12;
  const uint4* hp = (const uint4*)(hF + ((size_t)(bb << 10) + nn) * HD_ + hh * 64);
  float s = 0.f, dv = 0.f;
#pragma unroll
  for (int c = 0; c < 8; ++c) {
    union { uint4 u; __half h[8]; } cv; cv.u = hp[c];
#pragma unroll
    for (int q = 0; q < 8; ++q) {
      float v = __half2float(cv.h[q]);
      int d = c * 8 + q;
      s  += v * a_src[hh * 64 + d];
      dv += v * a_dst[hh * 64 + d];
    }
  }
  size_t o = ((size_t)(bb << 10) + nn) * H_ + hh;
  srcB[o] = s; dstB[o] = dv;
  if (mask[(bb << 10) + nn] > 0) atomicMaxF(&Mbuf[bb * H_ + hh], dv);
}

// -------- Kernel C: fused attention (streaming softmax + WMMA) + LayerNorm --------
#define A_OFF   0            // alphaS [4*64][136] f16 = 69632 B (aliased by outL [64][256] f32)
#define HT_OFF  69632        // hT [256][136] f16 = 69632 B
#define DST_OFF 139264       // dstF [4][1024] f32 = 16384 B
#define MSK_OFF 155648       // maskF [1024] i32 = 4096 B
#define INV_OFF 159744       // invL [4][64] f32 = 1024 B
#define SMEMC   160768

__global__ __launch_bounds__(256) void kC_attn(const __half* __restrict__ hF,
                                               const float* __restrict__ srcB,
                                               const float* __restrict__ dstB,
                                               const float* __restrict__ Mbuf,
                                               const int* __restrict__ adj,
                                               const int* __restrict__ mask,
                                               const float* __restrict__ gamma,
                                               const float* __restrict__ beta,
                                               float* __restrict__ out) {
  extern __shared__ __align__(16) char smemc[];
  __half* alphaS = (__half*)(smemc + A_OFF);
  __half* hT     = (__half*)(smemc + HT_OFF);
  float*  dstF   = (float*)(smemc + DST_OFF);
  int*    maskF  = (int*)(smemc + MSK_OFF);
  float*  invL   = (float*)(smemc + INV_OFF);
  float*  outL   = (float*)(smemc + A_OFF);

  int t = threadIdx.x;
  int i0 = blockIdx.x * 64;
  int bb = blockIdx.y;
  int w = t >> 5, lane = t & 31, l16 = lane & 15, lh = lane >> 4;
  int it = w >> 1, dh = w & 1;        // wave -> (i-tile 0..3, d-half 0..1)

  // stage dst scores + mask for the whole batch row range (once)
  for (int e = t; e < H_ * N_; e += 256) {
    int h = e >> 10, j = e & (N_ - 1);
    dstF[e] = dstB[((size_t)(bb << 10) + j) * H_ + h];
  }
  for (int j = t; j < N_; j += 256) maskF[j] = mask[(bb << 10) + j];
  __syncthreads();

  // each thread owns one (head, i) softmax row
  int head = t >> 6, irow = t & 63;
  float srcv = srcB[((size_t)(bb << 10) + i0 + irow) * H_ + head];
  float sm = srcv + Mbuf[bb * H_ + head];
  float mrow = sm > 0.f ? sm : 0.2f * sm;   // leaky-monotone upper bound of row max
  int mi = maskF[i0 + irow];
  const int4* arowBase = (const int4*)(adj + ((size_t)bb * N_ + i0 + irow) * N_);
  float l = 0.f;

  // hT staging: each thread owns a pair of adjacent j rows, 64 d values
  int jp = (t & 63) * 2, dbase2 = (t >> 6) * 64;
  v8f acc[8];
#pragma unroll
  for (int q = 0; q < 8; ++q)
#pragma unroll
    for (int e = 0; e < 8; ++e) acc[q][e] = 0.f;

  for (int jt = 0; jt < 8; ++jt) {
    int j0 = jt * 128;
    __syncthreads();                 // prior WMMA done with alphaS/hT
    { // transpose-stage h tile into hT[d][j]: j-pairs packed -> b32 stores
      const uint4* hrow0 = (const uint4*)(hF + ((size_t)(bb << 10) + j0 + jp)     * HD_ + dbase2);
      const uint4* hrow1 = (const uint4*)(hF + ((size_t)(bb << 10) + j0 + jp + 1) * HD_ + dbase2);
#pragma unroll
      for (int c = 0; c < 8; ++c) {
        union { uint4 u; __half h[8]; } c0, c1;
        c0.u = hrow0[c]; c1.u = hrow1[c];
        int d0 = dbase2 + c * 8;
#pragma unroll
        for (int q = 0; q < 8; ++q)
          *(__half2*)(hT + (d0 + q) * 136 + jp) = __halves2half2(c0.h[q], c1.h[q]);
      }
    }
    { // softmax row chunk: e = leaky(src+dst), mask, exp2, 4-wide packed f16 stores
      const int4* arow = arowBase + (j0 >> 2);
      if (jt < 7) __builtin_prefetch(arowBase + ((j0 + 128) >> 2), 0, 1);
      __half* aout = alphaS + (size_t)(head * 64 + irow) * 136;
      const float* dr = dstF + head * N_ + j0;
      const int* mr = maskF + j0;
      for (int c = 0; c < 32; ++c) {
        int4 av = arow[c];
        int jj = c * 4;
        float al[4];
#pragma unroll
        for (int u = 0; u < 4; ++u) {
          int a = (&av.x)[u];
          int j = jj + u;
          float ev = srcv + dr[j];
          ev = ev > 0.f ? ev : 0.2f * ev;
          bool valid = (a > 0) && (mr[j] > 0) && (mi > 0);
          al[u] = valid ? exp2f((ev - mrow) * LOG2E) : 0.f;
          l += al[u];
        }
        union { __half2 h2[2]; uint2 u2; } pk;
        pk.h2[0] = __halves2half2(__float2half(al[0]), __float2half(al[1]));
        pk.h2[1] = __halves2half2(__float2half(al[2]), __float2half(al[3]));
        *(uint2*)(aout + jj) = pk.u2;   // ds_store_b64
      }
    }
    __syncthreads();
    // WMMA: out[i, d] += alpha_h[i, j] * h[j, d]
#pragma unroll
    for (int ks = 0; ks < 4; ++ks) {
      int kb = ks * 32;
      v16h a0 = lds_a_frag(alphaS, 136, (dh * 2 + 0) * 64 + it * 16, kb);
      v16h a1 = lds_a_frag(alphaS, 136, (dh * 2 + 1) * 64 + it * 16, kb);
#pragma unroll
      for (int q = 0; q < 8; ++q) {
        v16h bf = lds_b_frag(hT, 136, dh * 128 + q * 16, kb);
        acc[q] = wmma16(q < 4 ? a0 : a1, bf, acc[q]);
      }
    }
  }
  __syncthreads();
  invL[head * 64 + irow] = (mi > 0 && l > 1e-30f) ? (1.f / l) : 0.f;
  __syncthreads();
  // scale by 1/l and stage into outL (aliases alphaS)
#pragma unroll
  for (int q = 0; q < 8; ++q) {
    int hq = (dh * 128 + q * 16) >> 6;
#pragma unroll
    for (int v = 0; v < 8; ++v) {
      int i = it * 16 + v + lh * 8;
      int d = dh * 128 + q * 16 + l16;
      outL[i * 256 + d] = acc[q][v] * invL[hq * 64 + i];
    }
  }
  __syncthreads();
  // LayerNorm, one row per wave iteration
  for (int r = 0; r < 8; ++r) {
    int i = w * 8 + r;
    float xs[8]; float s = 0.f, ss = 0.f;
#pragma unroll
    for (int k = 0; k < 8; ++k) {
      float xv = outL[i * 256 + lane + 32 * k];
      xs[k] = xv; s += xv; ss += xv * xv;
    }
#pragma unroll
    for (int off = 16; off > 0; off >>= 1) { s += __shfl_xor(s, off); ss += __shfl_xor(ss, off); }
    float mean = s * (1.f / 256.f);
    float var = ss * (1.f / 256.f) - mean * mean;
    float rstd = rsqrtf(var + 1e-5f);
    size_t ro = ((size_t)(bb << 10) + i0 + i) * (size_t)HD_;
#pragma unroll
    for (int k = 0; k < 8; ++k) {
      int d = lane + 32 * k;
      out[ro + d] = (xs[k] - mean) * rstd * gamma[d] + beta[d];
    }
  }
}

extern "C" void kernel_launch(void* const* d_in, const int* in_sizes, int n_in,
                              void* d_out, int out_size, void* d_ws, size_t ws_size,
                              hipStream_t stream) {
  const float* x     = (const float*)d_in[0];
  const int*   adj   = (const int*)d_in[1];
  const int*   mask  = (const int*)d_in[2];
  const float* Wg    = (const float*)d_in[3];
  const float* a_src = (const float*)d_in[4];
  const float* a_dst = (const float*)d_in[5];
  const float* gamma = (const float*)d_in[6];
  const float* beta  = (const float*)d_in[7];
  float* out = (float*)d_out;

  char* ws = (char*)d_ws;
  __half* hF  = (__half*)ws;                    // 16*1024*256 f16 = 8 MB
  float* srcB = (float*)(ws + 8388608);         // 256 KB
  float* dstB = (float*)(ws + 8650752);         // 256 KB
  float* Mbuf = (float*)(ws + 8912896);         // 64 floats

  hipFuncSetAttribute((const void*)kC_attn,
                      hipFuncAttributeMaxDynamicSharedMemorySize, SMEMC);

  kInit<<<1, 64, 0, stream>>>(Mbuf);
  kA_proj<<<dim3(2, 128), 256, 0, stream>>>(x, Wg, hF);
  kB_scores<<<256, 256, 0, stream>>>(hF, a_src, a_dst, mask, srcB, dstB, Mbuf);
  kC_attn<<<dim3(16, 16), 256, SMEMC, stream>>>(hF, srcB, dstB, Mbuf, adj, mask,
                                                gamma, beta, out);
}
